// TempSoftmaxDiagLinear_74689481277684
// MI455X (gfx1250) — matrix operand order repaired and checked by reading, not compiled, not run on hardware
//
#include <hip/hip_runtime.h>
#include <hip/hip_bf16.h>

// ---------------------------------------------------------------------------
// TempSoftmaxDiagLinear for MI455X (gfx1250, wave32, WMMA).
// out[8192,2048] = x[8192,2048] @ W^T,  W[r,c] = V[(r-c)&2047, c] * aw[(r-c)&2047]
// Strategy: alpha gate -> materialize W in split bf16 (hi+lo) -> bf16x3 WMMA GEMM
// W tiles are staged with GLOBAL_LOAD_ASYNC_TO_LDS_B128 when available.
// ---------------------------------------------------------------------------

#define IN_F   2048
#define OUT_F  2048
#define BATCH  8192
#define PDIM   2048
#define KGATE  1844.0f
#define TEMPER 1.0f

typedef __attribute__((ext_vector_type(16))) __bf16 v16bf;
typedef __attribute__((ext_vector_type(8)))  float  v8f;

union FragBF {           // one WMMA A/B operand: 16 bf16 = 8 VGPRs = 2x 16B chunks
    v16bf v;
    uint4 q[2];
};

#if __has_builtin(__builtin_amdgcn_global_load_async_to_lds_b128)
#define HAVE_ASYNC_LDS 1
// Parameter types per clang: v4i addrspace(1)* src, v4i addrspace(3)* dst, imm, imm
typedef int v4i_gcc __attribute__((vector_size(16)));
typedef __attribute__((address_space(1))) v4i_gcc* gbl_v4p;
typedef __attribute__((address_space(3))) v4i_gcc* lds_v4p;
#else
#define HAVE_ASYNC_LDS 0
#endif

#if HAVE_ASYNC_LDS
__device__ __forceinline__ void wait_asynccnt_zero() {
#if __has_builtin(__builtin_amdgcn_s_wait_asynccnt)
    __builtin_amdgcn_s_wait_asynccnt(0);
#else
    asm volatile("s_wait_asynccnt 0x0" ::: "memory");
#endif
}
#endif

__device__ __forceinline__ unsigned short f2bf(float f) {
    unsigned u = __builtin_bit_cast(unsigned, f);
    unsigned r = u + 0x7FFFu + ((u >> 16) & 1u);   // round-to-nearest-even
    return (unsigned short)(r >> 16);
}
__device__ __forceinline__ float bf2f(unsigned short h) {
    unsigned u = ((unsigned)h) << 16;
    return __builtin_bit_cast(float, u);
}

// ---------------------------------------------------------------------------
// Kernel 1: alpha_w = clip(K * softmax(alpha / T), 0, 1)      (P = 2048)
// ---------------------------------------------------------------------------
__global__ __launch_bounds__(256)
void alpha_gate_kernel(const float* __restrict__ alpha, float* __restrict__ aw) {
    __shared__ float red[256];
    const int t = threadIdx.x;

    float mx = -3.402823466e38f;
    for (int i = t; i < PDIM; i += 256) mx = fmaxf(mx, alpha[i] * (1.0f / TEMPER));
    red[t] = mx; __syncthreads();
    for (int s = 128; s > 0; s >>= 1) {
        if (t < s) red[t] = fmaxf(red[t], red[t + s]);
        __syncthreads();
    }
    mx = red[0]; __syncthreads();

    float sum = 0.0f;
    for (int i = t; i < PDIM; i += 256) sum += expf(alpha[i] * (1.0f / TEMPER) - mx);
    red[t] = sum; __syncthreads();
    for (int s = 128; s > 0; s >>= 1) {
        if (t < s) red[t] += red[t + s];
        __syncthreads();
    }
    sum = red[0];

    const float scale = KGATE / sum;
    for (int i = t; i < PDIM; i += 256) {
        float g = scale * expf(alpha[i] * (1.0f / TEMPER) - mx);
        aw[i] = fminf(1.0f, fmaxf(0.0f, g));
    }
}

// ---------------------------------------------------------------------------
// Kernel 2: materialize W (row-major [OUT_F][IN_F]) as split bf16 hi/lo.
// Each 64x64 output tile reads a *contiguous* 127-row band of V via LDS, so
// both the V gather and the W writes are fully coalesced.
// ---------------------------------------------------------------------------
__global__ __launch_bounds__(256)
void build_w_kernel(const float* __restrict__ V, const float* __restrict__ aw,
                    unsigned short* __restrict__ Wh, unsigned short* __restrict__ Wl) {
    __shared__ float sv[127 * 64];    // stride 64: anti-diagonal reads are conflict-free
    const int t  = threadIdx.x;
    const int c0 = blockIdx.x * 64;
    const int r0 = blockIdx.y * 64;
    const int pb = (r0 - c0 - 63 + 4096) & 2047;   // first diagonal index in band

    for (int f = t; f < 127 * 64; f += 256) {
        const int k = f >> 6, c = f & 63;
        const int p = (pb + k) & 2047;
        sv[f] = V[(size_t)p * IN_F + c0 + c] * aw[p];
    }
    __syncthreads();

    const int cl = t & 63;
    #pragma unroll
    for (int rg = 0; rg < 16; ++rg) {
        const int rl  = (t >> 6) + (rg << 2);
        const int idx = rl - cl + 63;              // in [0,126]
        const float w = sv[idx * 64 + cl];
        const unsigned short hb = f2bf(w);
        const unsigned short lb = f2bf(w - bf2f(hb));
        const size_t o = (size_t)(r0 + rl) * IN_F + (c0 + cl);
        Wh[o] = hb;
        Wl[o] = lb;
    }
}

// ---------------------------------------------------------------------------
// Kernel 3: GEMM via v_wmma_f32_16x16x32_bf16, bf16x3 emulated-f32 precision.
// Block tile 128x128x32, 8 waves, each wave: 2x4 grid of 16x16 C tiles.
// ---------------------------------------------------------------------------
#define BM  128
#define BN  128
#define BK  32
#define LDA 40          // LDS row stride in bf16 elems (80 B) to spread banks

__global__ __launch_bounds__(256)
void gemm_bf16x3_kernel(const float* __restrict__ X,
                        const unsigned short* __restrict__ Wh,
                        const unsigned short* __restrict__ Wl,
                        float* __restrict__ Out) {
    __shared__ __align__(16) unsigned short sAh[BM * LDA];
    __shared__ __align__(16) unsigned short sAl[BM * LDA];
    __shared__ __align__(16) unsigned short sBh[BN * LDA];
    __shared__ __align__(16) unsigned short sBl[BN * LDA];

    const int tid  = threadIdx.x;
    const int lane = tid & 31;
    const int wid  = tid >> 5;
    const int m0 = blockIdx.y * BM;     // batch dimension
    const int n0 = blockIdx.x * BN;     // out-feature dimension
    const int wm = (wid >> 1) * 32;     // wave row offset in tile
    const int wn = (wid & 1) * 64;      // wave col offset in tile
    const int mm = lane & 15;
    const int h  = lane >> 4;

    v8f acc[2][4];
    #pragma unroll
    for (int i = 0; i < 2; ++i)
        #pragma unroll
        for (int j = 0; j < 4; ++j)
            acc[i][j] = (v8f){0.f, 0.f, 0.f, 0.f, 0.f, 0.f, 0.f, 0.f};

    for (int kt = 0; kt < IN_F; kt += BK) {
        // ---- stage W tile (128 rows x 32 cols, bf16): pure copy ----
#if HAVE_ASYNC_LDS
        // Direct memory -> LDS, no VGPR round trip; tracked on ASYNCcnt.
        #pragma unroll
        for (int i = 0; i < 2; ++i) {
            const int f   = i * 256 + tid;      // 16B unit index [0,512)
            const int row = f >> 2;
            const int q   = (f & 3) << 3;       // bf16 col offset {0,8,16,24}
            __builtin_amdgcn_global_load_async_to_lds_b128(
                (gbl_v4p)(Wh + (size_t)(n0 + row) * IN_F + kt + q),
                (lds_v4p)&sBh[row * LDA + q], 0, 0);
            __builtin_amdgcn_global_load_async_to_lds_b128(
                (gbl_v4p)(Wl + (size_t)(n0 + row) * IN_F + kt + q),
                (lds_v4p)&sBl[row * LDA + q], 0, 0);
        }
#else
        #pragma unroll
        for (int i = 0; i < 2; ++i) {
            const int f   = i * 256 + tid;
            const int row = f >> 2;
            const int q   = (f & 3) << 3;
            *(uint4*)&sBh[row * LDA + q] =
                *(const uint4*)(Wh + (size_t)(n0 + row) * IN_F + kt + q);
            *(uint4*)&sBl[row * LDA + q] =
                *(const uint4*)(Wl + (size_t)(n0 + row) * IN_F + kt + q);
        }
#endif

        // ---- stage x tile (128x32 f32) -> split bf16 hi/lo in LDS (VALU path) ----
        #pragma unroll
        for (int i = 0; i < 4; ++i) {
            const int f   = i * 256 + tid;      // float4 unit index [0,1024)
            const int row = f >> 3;
            const int c4  = (f & 7) << 2;
            const float4 v = *(const float4*)(X + (size_t)(m0 + row) * IN_F + kt + c4);
            const unsigned short h0 = f2bf(v.x), h1 = f2bf(v.y),
                                 h2 = f2bf(v.z), h3 = f2bf(v.w);
            const unsigned short l0 = f2bf(v.x - bf2f(h0)), l1 = f2bf(v.y - bf2f(h1)),
                                 l2 = f2bf(v.z - bf2f(h2)), l3 = f2bf(v.w - bf2f(h3));
            *(uint2*)&sAh[row * LDA + c4] =
                make_uint2((unsigned)h0 | ((unsigned)h1 << 16),
                           (unsigned)h2 | ((unsigned)h3 << 16));
            *(uint2*)&sAl[row * LDA + c4] =
                make_uint2((unsigned)l0 | ((unsigned)l1 << 16),
                           (unsigned)l2 | ((unsigned)l3 << 16));
        }

        // prefetch next K tile while this one is consumed (global_prefetch_b8)
        if (kt + BK < IN_F) {
            __builtin_prefetch(X  + (size_t)(m0 + (tid >> 3)) * IN_F + (kt + BK) + ((tid & 7) << 2), 0, 3);
            __builtin_prefetch(Wh + (size_t)(n0 + (tid >> 2)) * IN_F + (kt + BK) + ((tid & 3) << 3), 0, 3);
            __builtin_prefetch(Wl + (size_t)(n0 + (tid >> 2)) * IN_F + (kt + BK) + ((tid & 3) << 3), 0, 3);
        }

#if HAVE_ASYNC_LDS
        wait_asynccnt_zero();    // drain this wave's async-to-LDS before barrier
#endif
        __syncthreads();

        // ---- load fragments ----
        // A (16x32, M x K): lane(m=mm, half h): K chunks [8h..8h+7] and [16+8h..16+8h+7]
        FragBF ah[2], al[2];
        #pragma unroll
        for (int i = 0; i < 2; ++i) {
            const unsigned short* pa = &sAh[(wm + 16 * i + mm) * LDA];
            ah[i].q[0] = *(const uint4*)(pa + 8 * h);
            ah[i].q[1] = *(const uint4*)(pa + 16 + 8 * h);
            const unsigned short* pl = &sAl[(wm + 16 * i + mm) * LDA];
            al[i].q[0] = *(const uint4*)(pl + 8 * h);
            al[i].q[1] = *(const uint4*)(pl + 16 + 8 * h);
        }
        // B (32x16, K x N) stored N-major (== W row-major): lane(n=mm, half h):
        // K chunks [8h..8h+7] and [16+8h..16+8h+7] of row n.
        FragBF bh[4], bl[4];
        #pragma unroll
        for (int j = 0; j < 4; ++j) {
            const unsigned short* pb = &sBh[(wn + 16 * j + mm) * LDA];
            bh[j].q[0] = *(const uint4*)(pb + 8 * h);
            bh[j].q[1] = *(const uint4*)(pb + 16 + 8 * h);
            const unsigned short* pl = &sBl[(wn + 16 * j + mm) * LDA];
            bl[j].q[0] = *(const uint4*)(pl + 8 * h);
            bl[j].q[1] = *(const uint4*)(pl + 16 + 8 * h);
        }

        // ---- bf16x3 WMMA: hi*hi + hi*lo + lo*hi, f32 accumulate ----
        #pragma unroll
        for (int i = 0; i < 2; ++i) {
            #pragma unroll
            for (int j = 0; j < 4; ++j) {
                acc[i][j] = __builtin_amdgcn_wmma_f32_16x16x32_bf16(
                    false, ah[i].v, false, bh[j].v, (short)0, acc[i][j], false, false);
                acc[i][j] = __builtin_amdgcn_wmma_f32_16x16x32_bf16(
                    false, ah[i].v, false, bl[j].v, (short)0, acc[i][j], false, false);
                acc[i][j] = __builtin_amdgcn_wmma_f32_16x16x32_bf16(
                    false, al[i].v, false, bh[j].v, (short)0, acc[i][j], false, false);
            }
        }
        __syncthreads();
    }

    // ---- epilogue: C layout lanes0-15 -> M=v, lanes16-31 -> M=8+v, N=mm ----
    #pragma unroll
    for (int i = 0; i < 2; ++i) {
        #pragma unroll
        for (int j = 0; j < 4; ++j) {
            float* o = Out + (size_t)(m0 + wm + 16 * i + 8 * h) * OUT_F
                           + (n0 + wn + 16 * j + mm);
            #pragma unroll
            for (int v = 0; v < 8; ++v)
                o[(size_t)v * OUT_F] = acc[i][j][v];
        }
    }
}

// ---------------------------------------------------------------------------
// Launch. Workspace layout: Wh (8 MB bf16) | Wl (8 MB bf16) | alpha_w (8 KB f32)
// ---------------------------------------------------------------------------
extern "C" void kernel_launch(void* const* d_in, const int* in_sizes, int n_in,
                              void* d_out, int out_size, void* d_ws, size_t ws_size,
                              hipStream_t stream) {
    const float* x     = (const float*)d_in[0];
    const float* V     = (const float*)d_in[1];
    const float* alpha = (const float*)d_in[2];
    // d_in[3]=rows_idx, d_in[4]=cols_idx: analytic diagonal form used instead.
    float* out = (float*)d_out;

    char* ws = (char*)d_ws;
    unsigned short* Wh = (unsigned short*)ws;
    unsigned short* Wl = Wh + (size_t)OUT_F * IN_F;
    float* aw = (float*)(ws + (size_t)OUT_F * IN_F * 2 * sizeof(unsigned short));

    alpha_gate_kernel<<<1, 256, 0, stream>>>(alpha, aw);
    build_w_kernel<<<dim3(IN_F / 64, OUT_F / 64), 256, 0, stream>>>(V, aw, Wh, Wl);
    gemm_bf16x3_kernel<<<dim3(OUT_F / BN, BATCH / BM), 256, 0, stream>>>(x, Wh, Wl, out);
}
